// MHABlock_64278480552303
// MI455X (gfx1250) — matrix operand order repaired and checked by reading, not compiled, not run on hardware
//
#include <hip/hip_runtime.h>
#include <hip/hip_bf16.h>
#include <stdint.h>

// ---------------------------------------------------------------------------
// MI455X (gfx1250) fused Swin window-attention block + GLU MLP.
// wave32, WMMA f32_16x16x32_f16 for every GEMM. All attention intermediates
// (qkv, scores, attn, attn-out) live in LDS; only x, y=x+attn, out touch HBM.
// ---------------------------------------------------------------------------

typedef __attribute__((ext_vector_type(16))) _Float16 v16h;
typedef __attribute__((ext_vector_type(8)))  float    v8f;

#define WS_   8
#define NH_   6
#define C_    192
#define HD_   32
#define N_    64          // tokens per window
#define B_    8
#define H_    128
#define W_    128
#define HW_   (H_ * W_)   // 16384
#define NWIN  2048        // 8 * 16 * 16
#define QKV_SCALE 0.17677669529663687f  // 1/sqrt(32)

union F8 { v8f v; float f[8]; };

// ---- A fragment: matrix stored row-major [M][K], stride in halves.
// Lane L: row = row0 + (L&15); halves 0..7 = K[k0 + hi*8 ..], 8..15 = K[k0+16+hi*8 ..]
__device__ __forceinline__ v16h frag_a(const _Float16* base, int row0, int k0, int stride) {
  const int lane = threadIdx.x & 31;
  const _Float16* p = base + (row0 + (lane & 15)) * stride + k0 + ((lane >> 4) << 3);
  union { v16h v; uint4 u[2]; } t;
  t.u[0] = *(const uint4*)(p);
  t.u[1] = *(const uint4*)(p + 16);
  return t.v;
}

// ---- B fragment: operand stored as B^T row-major [N][K] (e.g. weights [o][c]).
// Lane L: col n = n0 + (L&15); halves 0..7 = K[k0 + hi*8 ..], 8..15 = K[k0+16+hi*8 ..]
__device__ __forceinline__ v16h frag_b(const _Float16* base, int n0, int k0, int stride) {
  const int lane = threadIdx.x & 31;
  const _Float16* p = base + (n0 + (lane & 15)) * stride + k0 + ((lane >> 4) << 3);
  union { v16h v; uint4 u[2]; } t;
  t.u[0] = *(const uint4*)(p);
  t.u[1] = *(const uint4*)(p + 16);
  return t.v;
}

__device__ __forceinline__ v8f wmma_f16(v16h a, v16h b, v8f c) {
  return __builtin_amdgcn_wmma_f32_16x16x32_f16(false, a, false, b, (short)0, c, false, false);
}

// ---------------------------------------------------------------------------
// Kernel 0: fp32 -> f16 weight conversion (layout preserved: [o][c] == B^T).
// ---------------------------------------------------------------------------
__global__ void cvt_f16_kernel(const float* __restrict__ src,
                               _Float16* __restrict__ dst, int n) {
  int i = blockIdx.x * blockDim.x + threadIdx.x;
  int stride = gridDim.x * blockDim.x;
  for (; i < n; i += stride) dst[i] = (_Float16)src[i];
}

// ---------------------------------------------------------------------------
// Kernel 1: per-window fused  QKV -> attention(+bias, softmax) -> proj -> +x
// block = 192 threads (6 waves, one head per wave), 160 KB dynamic LDS.
// LDS map (halves unless noted):
//   sX    [64][192]   24 KB   window x (f16)  -> reused as attn-out after QKV
//   sQ    [6][64][32] 24 KB   Q (pre-scaled)
//   sK    [6][64][32] 24 KB   K  ([tok][d] == B^T for QK^T)
//   sVt   [6][32][64] 24 KB   V^T ([d][tok] == B^T for attn@V)
//   sB    [64][64]f32 16 KB   relative-position bias, gathered once
//   sAttn [6][64][64] 48 KB   softmax(A) as f16 (A operand of attn@V)
// ---------------------------------------------------------------------------
__global__ __launch_bounds__(192) void attn_block_kernel(
    const float* __restrict__ x,
    const _Float16* __restrict__ wqkv,   // [576][192] f16
    const float* __restrict__ qkv_b,     // [576]
    const _Float16* __restrict__ wproj,  // [192][192] f16
    const float* __restrict__ proj_b,    // [192]
    const float* __restrict__ bias_table,
    const int* __restrict__ rel_index,   // [64*64]
    float* __restrict__ y) {

  extern __shared__ char smem[];
  _Float16* sX   = (_Float16*)smem;                 // 12288 halves
  _Float16* sQ   = sX + 64 * 192;                   // 12288
  _Float16* sK   = sQ + NH_ * N_ * HD_;             // 12288
  _Float16* sVt  = sK + NH_ * N_ * HD_;             // 12288
  float*    sB   = (float*)(sVt + NH_ * HD_ * N_);  // 4096 f32
  _Float16* sAttn = (_Float16*)(sB + N_ * N_);      // 24576 halves

  const int tid  = threadIdx.x;
  const int wave = tid >> 5;       // 0..5  == head id
  const int lane = tid & 31;
  const int nc   = lane & 15;
  const int hi   = lane >> 4;

  const int win  = blockIdx.x;
  const int bidx = win >> 8;                 // 256 windows per batch image
  const int wr   = (win >> 4) & 15;
  const int wc   = win & 15;
  const int h0   = wr * WS_;
  const int w0   = wc * WS_;

  // hint weight stream into cache (global_prefetch_b8)
  __builtin_prefetch(wqkv + tid * 64, 0, 1);
  __builtin_prefetch(wproj + tid * 32, 0, 1);

  // ---- load window x -> f16 LDS ([tok][c]), gather bias table -> LDS
  for (int i = tid; i < N_ * C_; i += 192) {
    int c = i >> 6, m = i & 63;
    int hh = h0 + (m >> 3), ww = w0 + (m & 7);
    sX[m * C_ + c] = (_Float16)x[(((size_t)bidx * C_ + c) * H_ + hh) * W_ + ww];
  }
  for (int i = tid; i < N_ * N_; i += 192)
    sB[i] = bias_table[rel_index[i]];
  __syncthreads();

  // ---- QKV GEMM: [64,192] x [192,576]
  // Part p (0=Q,1=K,2=V) is a compile-time-unrolled loop so each store block
  // is straight-line (no per-element EXEC divergence); wave owns col tiles
  // p*12 + wave*2 + {0,1}.
#pragma unroll
  for (int p = 0; p < 3; ++p) {
    for (int t = 0; t < 2; ++t) {
      const int ct = wave * 2 + t;               // col tile within part, 0..11
      const int o0 = p * C_ + ct * 16;           // global output col tile base
      const int cc = ct * 16 + nc;               // col within part, 0..191
      const int head = cc >> 5, d = cc & 31;
      const float bb = qkv_b[o0 + nc];
      for (int tm = 0; tm < 4; ++tm) {
        F8 acc = {};
#pragma unroll
        for (int kk = 0; kk < 6; ++kk)
          acc.v = wmma_f16(frag_a(sX, tm * 16, kk * 32, C_),
                           frag_b(wqkv, o0, kk * 32, C_), acc.v);
        const int m0 = tm * 16 + (hi << 3);      // first of 8 rows this lane
        if (p == 0) {
          _Float16* dst = sQ + head * (N_ * HD_) + m0 * HD_ + d;
#pragma unroll
          for (int r = 0; r < 8; ++r)
            dst[r * HD_] = (_Float16)((acc.f[r] + bb) * QKV_SCALE);
        } else if (p == 1) {
          _Float16* dst = sK + head * (N_ * HD_) + m0 * HD_ + d;
#pragma unroll
          for (int r = 0; r < 8; ++r)
            dst[r * HD_] = (_Float16)(acc.f[r] + bb);
        } else {
          _Float16* dst = sVt + head * (HD_ * N_) + d * N_ + m0;
#pragma unroll
          for (int r = 0; r < 8; ++r)
            dst[r] = (_Float16)(acc.f[r] + bb);
        }
      }
    }
  }
  __syncthreads();

  // ---- per-head attention (head == wave). HD=32 => one k-step per QK tile.
  {
    const _Float16* Qh = sQ  + wave * (N_ * HD_);
    const _Float16* Kh = sK  + wave * (N_ * HD_);
    const _Float16* Vh = sVt + wave * (HD_ * N_);
    _Float16*       Ah = sAttn + wave * (N_ * N_);

    for (int tm = 0; tm < 4; ++tm) {
      // scores row-block [16 x 64]
      F8 sc[4];
      const v16h aq = frag_a(Qh, tm * 16, 0, HD_);
#pragma unroll
      for (int tn = 0; tn < 4; ++tn) {
        v8f z = {};
        sc[tn].v = wmma_f16(aq, frag_b(Kh, tn * 16, 0, HD_), z);
      }
      // bias + numerically stable softmax over the 64 columns
#pragma unroll
      for (int r = 0; r < 8; ++r) {
        const int m = tm * 16 + r + (hi << 3);
        float v0 = sc[0].f[r] + sB[m * 64 + nc];
        float v1 = sc[1].f[r] + sB[m * 64 + 16 + nc];
        float v2 = sc[2].f[r] + sB[m * 64 + 32 + nc];
        float v3 = sc[3].f[r] + sB[m * 64 + 48 + nc];
        float mx = fmaxf(fmaxf(v0, v1), fmaxf(v2, v3));
#pragma unroll
        for (int msk = 8; msk >= 1; msk >>= 1)
          mx = fmaxf(mx, __shfl_xor(mx, msk, 32));   // within 16-lane half
        v0 = __expf(v0 - mx); v1 = __expf(v1 - mx);
        v2 = __expf(v2 - mx); v3 = __expf(v3 - mx);
        float sm = v0 + v1 + v2 + v3;
#pragma unroll
        for (int msk = 8; msk >= 1; msk >>= 1)
          sm += __shfl_xor(sm, msk, 32);
        const float inv = 1.0f / sm;
        Ah[m * 64 + nc]      = (_Float16)(v0 * inv);
        Ah[m * 64 + 16 + nc] = (_Float16)(v1 * inv);
        Ah[m * 64 + 32 + nc] = (_Float16)(v2 * inv);
        Ah[m * 64 + 48 + nc] = (_Float16)(v3 * inv);
      }
      // attn @ V : [16,64] x [64,32]
#pragma unroll
      for (int td = 0; td < 2; ++td) {
        F8 o = {};
#pragma unroll
        for (int kk = 0; kk < 2; ++kk)
          o.v = wmma_f16(frag_a(Ah, tm * 16, kk * 32, N_),
                         frag_b(Vh, td * 16, kk * 32, N_), o.v);
        // attn-out reuses sX: [tok][192]
        _Float16* dst = sX + (tm * 16 + (hi << 3)) * C_ + wave * HD_ + td * 16 + nc;
#pragma unroll
        for (int r = 0; r < 8; ++r)
          dst[r * C_] = (_Float16)o.f[r];
      }
    }
  }
  __syncthreads();

  // ---- proj GEMM + residual: y = x + attn_out @ Wp^T + bp
  for (int t = 0; t < 2; ++t) {
    const int c0 = (wave * 2 + t) * 16;
    const int cc = c0 + nc;
    const float pb = proj_b[cc];
    for (int tm = 0; tm < 4; ++tm) {
      F8 acc = {};
#pragma unroll
      for (int kk = 0; kk < 6; ++kk)
        acc.v = wmma_f16(frag_a(sX, tm * 16, kk * 32, C_),
                         frag_b(wproj, c0, kk * 32, C_), acc.v);
#pragma unroll
      for (int r = 0; r < 8; ++r) {
        const int m = tm * 16 + r + (hi << 3);
        const size_t g = (((size_t)bidx * C_ + cc) * H_ + h0 + (m >> 3)) * W_ + w0 + (m & 7);
        y[g] = x[g] + acc.f[r] + pb;
      }
    }
  }
}

// ---------------------------------------------------------------------------
// Kernel 2: fused GLU MLP over 64-pixel tiles.
// block = 384 threads (12 waves). Wave w owns value col-tile w and gate
// col-tile 12+w of GEMM1 so GLU pairing is register-local; GEMM2 col-tile w.
// ---------------------------------------------------------------------------
__global__ __launch_bounds__(384) void glu_mlp_kernel(
    const float* __restrict__ y,
    const _Float16* __restrict__ w1,   // [384][192]
    const float* __restrict__ b1,      // [384]
    const _Float16* __restrict__ w2,   // [192][192]
    const float* __restrict__ b2,      // [192]
    float* __restrict__ out) {

  __shared__ _Float16 sT[64 * C_];     // 24 KB: y tile (f16), reused for g

  const int tid  = threadIdx.x;
  const int wave = tid >> 5;           // 0..11
  const int lane = tid & 31;
  const int nc   = lane & 15;
  const int hi   = lane >> 4;

  const int pix0 = blockIdx.x * 64;
  const int bb   = pix0 >> 14;         // / 16384
  const int hw0  = pix0 & (HW_ - 1);

  __builtin_prefetch(w1 + tid * 64, 0, 1);
  __builtin_prefetch(w2 + tid * 32, 0, 1);

  // load y tile -> f16 LDS [pix][c] (coalesced over pixels)
  for (int i = tid; i < 64 * C_; i += 384) {
    int c = i >> 6, p = i & 63;
    sT[p * C_ + c] = (_Float16)y[((size_t)bb * C_ + c) * HW_ + hw0 + p];
  }
  __syncthreads();

  // GEMM1 (both halves), keep h in registers
  F8 hA[4], hB[4];
  for (int tm = 0; tm < 4; ++tm) {
    F8 ca = {}, cb = {};
#pragma unroll
    for (int kk = 0; kk < 6; ++kk) {
      const v16h a = frag_a(sT, tm * 16, kk * 32, C_);
      ca.v = wmma_f16(a, frag_b(w1, wave * 16,      kk * 32, C_), ca.v);
      cb.v = wmma_f16(a, frag_b(w1, C_ + wave * 16, kk * 32, C_), cb.v);
    }
    hA[tm] = ca; hB[tm] = cb;
  }
  __syncthreads();                      // all reads of sT done

  // GLU gate, write g (f16) back into sT
  const float ba  = b1[wave * 16 + nc];
  const float bbv = b1[C_ + wave * 16 + nc];
#pragma unroll
  for (int tm = 0; tm < 4; ++tm) {
    _Float16* dst = sT + (tm * 16 + (hi << 3)) * C_ + wave * 16 + nc;
#pragma unroll
    for (int r = 0; r < 8; ++r) {
      const float av = hA[tm].f[r] + ba;
      const float bv = hB[tm].f[r] + bbv;
      dst[r * C_] = (_Float16)(av * (1.0f / (1.0f + __expf(-bv))));
    }
  }
  __syncthreads();

  // GEMM2 + residual: out = y + g @ W2^T + b2
  const int cc = wave * 16 + nc;
  const float bv2 = b2[cc];
  for (int tm = 0; tm < 4; ++tm) {
    F8 acc = {};
#pragma unroll
    for (int kk = 0; kk < 6; ++kk)
      acc.v = wmma_f16(frag_a(sT, tm * 16, kk * 32, C_),
                       frag_b(w2, wave * 16, kk * 32, C_), acc.v);
#pragma unroll
    for (int r = 0; r < 8; ++r) {
      const int m = tm * 16 + r + (hi << 3);
      const size_t g = ((size_t)bb * C_ + cc) * HW_ + hw0 + m;
      out[g] = y[g] + acc.f[r] + bv2;
    }
  }
}

// ---------------------------------------------------------------------------
extern "C" void kernel_launch(void* const* d_in, const int* in_sizes, int n_in,
                              void* d_out, int out_size, void* d_ws, size_t ws_size,
                              hipStream_t stream) {
  const float* x          = (const float*)d_in[0];
  const float* qkv_w      = (const float*)d_in[1];   // [576][192]
  const float* qkv_b      = (const float*)d_in[2];
  const float* proj_w     = (const float*)d_in[3];   // [192][192]
  const float* proj_b     = (const float*)d_in[4];
  const float* mlp1_w     = (const float*)d_in[5];   // [384][192]
  const float* mlp1_b     = (const float*)d_in[6];
  const float* mlp2_w     = (const float*)d_in[7];   // [192][192]
  const float* mlp2_b     = (const float*)d_in[8];
  const float* bias_table = (const float*)d_in[9];
  const int*   rel_index  = (const int*)d_in[10];
  float* out = (float*)d_out;

  // workspace layout
  uintptr_t p = (uintptr_t)d_ws;
  _Float16* wqkv_h = (_Float16*)p; p += (size_t)(3 * C_ * C_) * sizeof(_Float16);
  _Float16* wproj_h = (_Float16*)p; p += (size_t)(C_ * C_) * sizeof(_Float16);
  _Float16* w1_h = (_Float16*)p; p += (size_t)(2 * C_ * C_) * sizeof(_Float16);
  _Float16* w2_h = (_Float16*)p; p += (size_t)(C_ * C_) * sizeof(_Float16);
  p = (p + 255) & ~(uintptr_t)255;
  float* y = (float*)p;                               // [8,192,128,128] fp32

  // weight conversion (tiny; recomputed every call -> deterministic)
  cvt_f16_kernel<<<(3 * C_ * C_ + 255) / 256, 256, 0, stream>>>(qkv_w,  wqkv_h,  3 * C_ * C_);
  cvt_f16_kernel<<<(C_ * C_ + 255) / 256,     256, 0, stream>>>(proj_w, wproj_h, C_ * C_);
  cvt_f16_kernel<<<(2 * C_ * C_ + 255) / 256, 256, 0, stream>>>(mlp1_w, w1_h,    2 * C_ * C_);
  cvt_f16_kernel<<<(C_ * C_ + 255) / 256,     256, 0, stream>>>(mlp2_w, w2_h,    C_ * C_);

  // fused window attention (+residual1) -> y
  const size_t smem_attn = 160 * 1024;   // 24+24+24+24+16+48 KB
  attn_block_kernel<<<NWIN, 192, smem_attn, stream>>>(
      x, wqkv_h, qkv_b, wproj_h, proj_b, bias_table, rel_index, y);

  // fused GLU MLP (+residual2) -> out
  glu_mlp_kernel<<<(B_ * HW_) / 64, 384, 0, stream>>>(y, w1_h, mlp1_b, w2_h, mlp2_b, out);
}
